// Oscillator_RNN_dyn_90237262889225
// MI455X (gfx1250) — matrix operand order repaired and checked by reading, not compiled, not run on hardware
//
#include <hip/hip_runtime.h>
#include <hip/hip_bf16.h>

// ---------------- problem constants (from reference) ----------------
#define N_IN   784
#define N_OUT  10
#define NN     1024
#define BB     32
#define ALPHA_F  0.9f
#define EPS_INT  0.05f
#define DT_F     0.05f
#define MAX_STEPS 1000

// ---------------- launch shape ----------------
#define NBLK 32
#define NTHR 256
#define NTHREADS (NBLK * NTHR)          // 8192
#define WPB (NTHR / 32)                 // 8 waves / block
// 256 waves total: 128 output tiles per GEMM (2 x 64 tiles of 16x16)

typedef __attribute__((ext_vector_type(16))) __bf16 v16bf;
typedef __attribute__((ext_vector_type(8)))  __bf16 v8bf;
typedef __attribute__((ext_vector_type(8)))  float  v8f;

// ---------------- workspace layout (bytes, 256-aligned) ----------------
static constexpr size_t OFF_XIN   = 0;          // f32 [32][1024]  131072
static constexpr size_t OFF_H0    = 131072;     // f32 [32][1024]
static constexpr size_t OFF_H1    = 262144;     // f32 [32][1024]
static constexpr size_t OFF_S1    = 393216;     // f32 [32][1024]  (h1 @ Wrec1^T)
static constexpr size_t OFF_H0BF0 = 524288;     // bf16 [32][1024] 65536
static constexpr size_t OFF_H0BF1 = 589824;     // bf16 [32][1024]
static constexpr size_t OFF_H1BF  = 655360;     // bf16 [32][1024]
static constexpr size_t OFF_WM0   = 720896;     // bf16 [1024][1024] 2097152
static constexpr size_t OFF_WM1   = 2818048;    // bf16 [1024][1024]
static constexpr size_t OFF_WI1   = 4915200;    // bf16 [1024][1024]
static constexpr size_t OFF_CTRL  = 7012352;    // control block

struct Ctrl {
    unsigned cnt;       // barrier arrival counter
    unsigned gen;       // barrier generation
    unsigned delta[2];  // max|dh| per iteration parity (float bits, >=0)
};

// ---------------- software grid barrier (all NBLK blocks co-resident) ----
__device__ __forceinline__ void grid_barrier(Ctrl* c) {
    __syncthreads();
    if (threadIdx.x == 0) {
        __threadfence();                      // publish prior writes device-wide
        volatile unsigned* genv = &c->gen;
        unsigned g = *genv;
        unsigned a = atomicAdd(&c->cnt, 1u);
        if (a == (unsigned)(NBLK - 1)) {
            *(volatile unsigned*)&c->cnt = 0u;
            __threadfence();
            atomicAdd(&c->gen, 1u);
        } else {
            while (*genv == g) { __builtin_amdgcn_s_sleep(1); }
        }
        __threadfence();                      // acquire
    }
    __syncthreads();
}

// ---------------- one 16x16 output tile of [32x1024]x[1024x1024]^T -------
// A: hbf row-major [32][1024] bf16 (M x K tile 16x32)
// W: row-major [N][K]=[1024][1024] bf16 ; out[m,n] = sum_k A[m,k]*W[n,k]
// B-fragment (K x N = 32x16): lanes 0-15 hold K=0..15 of column n=lane,
// lanes 16-31 hold K=16..31  -> one contiguous 32B run per lane from W.
// A-fragment (16x32): lanes 0-15: K 0..15; lanes 16-31: K 16..31.
__device__ __forceinline__ v8f wmma_tile(const __bf16* __restrict__ hbf,
                                         const __bf16* __restrict__ W,
                                         int mt, int nt, int lane) {
    v8f acc = {};
    const int ml  = lane & 15;
    const int hiH = (lane & 16) ? 16 : 0;   // A half-wave K shift
    const int hiB = (lane & 16) ? 16 : 0;   // B half-wave K shift
    const __bf16* ap = hbf + (size_t)(mt * 16 + ml) * NN + hiH;
    const __bf16* bp = W   + (size_t)(nt * 16 + ml) * NN + hiB;
#pragma unroll 4
    for (int kk = 0; kk < NN / 32; ++kk) {
        v8bf alo = *(const v8bf*)(ap);        // K +0..7   (16B)
        v8bf ahi = *(const v8bf*)(ap + 8);    // K +8..15  (16B)
        v16bf a = __builtin_shufflevector(alo, ahi,
                    0,1,2,3,4,5,6,7,8,9,10,11,12,13,14,15);
        v16bf b = *(const v16bf*)(bp);        // 32B contiguous
        acc = __builtin_amdgcn_wmma_f32_16x16x32_bf16(
                  false, a, false, b, (short)0, acc, false, false);
        ap += 32; bp += 32;
    }
    return acc;
}

__device__ __forceinline__ void wave_max_atomic(float v, unsigned* slot, int lane) {
#pragma unroll
    for (int off = 16; off > 0; off >>= 1)
        v = fmaxf(v, __shfl_xor(v, off, 32));
    if (lane == 0) atomicMax(slot, __float_as_uint(v));   // v>=0: uint order == float order
}

// ---------------- init kernel: reset control block every call ------------
__global__ void osc_init_kernel(unsigned char* ws) {
    Ctrl* c = (Ctrl*)(ws + OFF_CTRL);
    c->cnt = 0u; c->gen = 0u; c->delta[0] = 0u; c->delta[1] = 0u;
}

// ---------------- persistent kernel --------------------------------------
__global__ void __launch_bounds__(NTHR, 1)
osc_rnn_kernel(const float* __restrict__ X,      const float* __restrict__ W_input,
               const float* __restrict__ b_input,const float* __restrict__ W_in,
               const float* __restrict__ b_in,   const float* __restrict__ W_rec,
               const float* __restrict__ mask,   const float* __restrict__ b_rec,
               const float* __restrict__ W_out,  const float* __restrict__ b_out,
               float* __restrict__ out, unsigned char* __restrict__ ws)
{
    float*  x_in  = (float*)(ws + OFF_XIN);
    float*  h0    = (float*)(ws + OFF_H0);
    float*  h1    = (float*)(ws + OFF_H1);
    float*  s1    = (float*)(ws + OFF_S1);
    __bf16* h0bf0 = (__bf16*)(ws + OFF_H0BF0);
    __bf16* h0bf1 = (__bf16*)(ws + OFF_H0BF1);
    __bf16* h1bf  = (__bf16*)(ws + OFF_H1BF);
    __bf16* Wm0   = (__bf16*)(ws + OFF_WM0);
    __bf16* Wm1   = (__bf16*)(ws + OFF_WM1);
    __bf16* Wi1   = (__bf16*)(ws + OFF_WI1);
    Ctrl*   ctrl  = (Ctrl*)(ws + OFF_CTRL);

    const int tid  = blockIdx.x * NTHR + threadIdx.x;
    const int lane = threadIdx.x & 31;
    const int w    = blockIdx.x * WPB + (threadIdx.x >> 5);   // 0..255

    // ---------------- setup: x_in (f32), bf16 weights, zero state --------
    for (int e = tid; e < BB * NN; e += NTHREADS) {
        const int m = e >> 10, n = e & (NN - 1);
        const float* xr = X + (size_t)m * N_IN;
        const float* wr = W_input + (size_t)n * N_IN;
        float s = b_input[n];
        for (int j = 0; j < N_IN; ++j) s = fmaf(xr[j], wr[j], s);
        x_in[e] = s;
        h0[e] = 0.f; h1[e] = 0.f;
        h0bf0[e] = (__bf16)0.f; h0bf1[e] = (__bf16)0.f; h1bf[e] = (__bf16)0.f;
    }
    for (int i = tid; i < NN * NN; i += NTHREADS) {
        Wm0[i] = (__bf16)(W_rec[i] * mask[i]);
        Wm1[i] = (__bf16)(W_rec[(size_t)NN * NN + i] * mask[(size_t)NN * NN + i]);
        Wi1[i] = (__bf16)(W_in[(size_t)NN * NN + i]);
    }
    grid_barrier(ctrl);

    const float* brec0 = b_rec;
    const float* brec1 = b_rec + NN;
    const float* bin1  = b_in + NN;

    // ---------------- fixed-point iteration ------------------------------
    for (int k = 0; k < MAX_STEPS; ++k) {
        const int p = k & 1;
        const __bf16* h0bf_r = p ? h0bf1 : h0bf0;   // current h0 (bf16)
        __bf16*       h0bf_w = p ? h0bf0 : h0bf1;   // new h0 (bf16)

        // Phase A: layer-0 update (waves 0..127) || S1 = h1 @ Wm1^T (128..255)
        if (w < 128) {
            const int mt = w >> 6, nt = w & 63;
            v8f acc = wmma_tile(h0bf_r, Wm0, mt, nt, lane);
            const int nc = nt * 16 + (lane & 15);
            const int m0 = mt * 16 + ((lane & 16) ? 8 : 0);
            float lmax = 0.f;
#pragma unroll
            for (int r = 0; r < 8; ++r) {
                const int idx = (m0 + r) * NN + nc;
                const float u  = x_in[idx] + acc[r] + brec0[nc];
                const float h  = h0[idx];
                const float dh = fmaf(-ALPHA_F, h, sinf(u));
                const float hn = fmaf(DT_F, dh, h);
                h0[idx] = hn;
                h0bf_w[idx] = (__bf16)hn;
                lmax = fmaxf(lmax, fabsf(dh));
            }
            wave_max_atomic(lmax, &ctrl->delta[p], lane);
        } else {
            const int t = w - 128, mt = t >> 6, nt = t & 63;
            v8f acc = wmma_tile(h1bf, Wm1, mt, nt, lane);
            const int nc = nt * 16 + (lane & 15);
            const int m0 = mt * 16 + ((lane & 16) ? 8 : 0);
#pragma unroll
            for (int r = 0; r < 8; ++r) s1[(m0 + r) * NN + nc] = acc[r];
        }
        grid_barrier(ctrl);

        // Phase B: layer-1 update using new_h0 (waves 0..127)
        if (w < 128) {
            const int mt = w >> 6, nt = w & 63;
            v8f acc = wmma_tile(h0bf_w, Wi1, mt, nt, lane);
            const int nc = nt * 16 + (lane & 15);
            const int m0 = mt * 16 + ((lane & 16) ? 8 : 0);
            float lmax = 0.f;
#pragma unroll
            for (int r = 0; r < 8; ++r) {
                const int idx = (m0 + r) * NN + nc;
                const float u  = acc[r] + bin1[nc] + s1[idx] + brec1[nc];
                const float h  = h1[idx];
                const float dh = fmaf(-ALPHA_F, h, sinf(u));
                const float hn = fmaf(DT_F, dh, h);
                h1[idx] = hn;
                h1bf[idx] = (__bf16)hn;
                lmax = fmaxf(lmax, fabsf(dh));
            }
            wave_max_atomic(lmax, &ctrl->delta[p], lane);
        }
        if (tid == 0) *(volatile unsigned*)&ctrl->delta[p ^ 1] = 0u; // slot for iter k+1
        grid_barrier(ctrl);

        const unsigned mdbits = *(volatile unsigned*)&ctrl->delta[p];
        if (__uint_as_float(mdbits) < EPS_INT) break;   // uniform across all waves
    }

    // ---------------- output head: out = h1 @ W_out^T + b_out (f32) ------
    if (tid < BB * N_OUT) {
        const int m = tid / N_OUT, o = tid % N_OUT;
        const float* hr = h1 + (size_t)m * NN;
        const float* wr = W_out + (size_t)o * NN;
        float s = b_out[o];
        for (int j = 0; j < NN; ++j) s = fmaf(hr[j], wr[j], s);
        out[tid] = s;
    }
}

// ---------------- entry point --------------------------------------------
extern "C" void kernel_launch(void* const* d_in, const int* in_sizes, int n_in,
                              void* d_out, int out_size, void* d_ws, size_t ws_size,
                              hipStream_t stream) {
    (void)in_sizes; (void)n_in; (void)out_size; (void)ws_size;
    const float* X       = (const float*)d_in[0];
    const float* W_input = (const float*)d_in[1];
    const float* b_input = (const float*)d_in[2];
    const float* W_in    = (const float*)d_in[3];
    const float* b_in    = (const float*)d_in[4];
    const float* W_rec   = (const float*)d_in[5];
    const float* mask    = (const float*)d_in[6];
    const float* b_rec   = (const float*)d_in[7];
    const float* W_out   = (const float*)d_in[8];
    const float* b_out   = (const float*)d_in[9];

    unsigned char* ws = (unsigned char*)d_ws;
    osc_init_kernel<<<1, 1, 0, stream>>>(ws);
    osc_rnn_kernel<<<NBLK, NTHR, 0, stream>>>(X, W_input, b_input, W_in, b_in,
                                              W_rec, mask, b_rec, W_out, b_out,
                                              (float*)d_out, ws);
}
